// RoPEAttention_42691974922557
// MI455X (gfx1250) — matrix operand order repaired and checked by reading, not compiled, not run on hardware
//
#include <hip/hip_runtime.h>
#include <hip/hip_bf16.h>
#include <stdint.h>

// ---------------------------------------------------------------------------
// MI455X (gfx1250) RoPE attention, bf16 WMMA pipeline with async global->LDS
// staging (ASYNCcnt path).  B=2, N=2048, C=1024, H=16, hd=64.
// ---------------------------------------------------------------------------

typedef __bf16 v16bf __attribute__((ext_vector_type(16)));
typedef float  v8f   __attribute__((ext_vector_type(8)));

__device__ __forceinline__ unsigned short f2bf(float f) {
  union { float f; uint32_t u; } x; x.f = f;
  uint32_t r = x.u + 0x7FFFu + ((x.u >> 16) & 1u);   // round-to-nearest-even
  return (unsigned short)(r >> 16);
}

union FragU { v16bf v; uint4 q[2]; };

// Async copy: 16B per lane, global -> LDS, tracked by ASYNCcnt.
// VDST carries the wave-relative LDS byte address (= low 32 bits of the
// generic pointer to __shared__ memory).
__device__ __forceinline__ void async_copy_b128(const void* gsrc, void* ldst) {
  unsigned loff = (unsigned)(uintptr_t)ldst;
  asm volatile("global_load_async_to_lds_b128 %0, %1, off"
               :: "v"(loff), "v"(gsrc) : "memory");
}
__device__ __forceinline__ void wait_async0() {
  asm volatile("s_wait_asynccnt 0" ::: "memory");
}
__device__ __forceinline__ void wait_ds0() {
  asm volatile("s_wait_dscnt 0" ::: "memory");
}

// A-matrix fragment, 16x32 bf16, row-major source [.., lda]
// lanes 0-15: M=lane, K 0..15; lanes 16-31: M=lane-16, K 16..31
__device__ __forceinline__ v16bf load_A_frag(const unsigned short* base, int lda,
                                             int row0, int k0) {
  int lane = threadIdx.x & 31;
  int hi = lane >> 4, r = lane & 15;
  const unsigned short* p = base + (size_t)(row0 + r) * lda + k0 + hi * 16;
  FragU u;
  u.q[0] = *(const uint4*)p;
  u.q[1] = *(const uint4*)(p + 8);
  return u.v;
}

// B-matrix fragment, 32x16 bf16 where B(k,n) = W[n][k]; W row-major [.., ldw]
// lanes 0-15: N=lane, K 0..15; lanes 16-31: N=lane-16, K 16..31
__device__ __forceinline__ v16bf load_B_frag(const unsigned short* W, int ldw,
                                             int n0, int k0) {
  int lane = threadIdx.x & 31;
  int hi = lane >> 4, c = lane & 15;
  const unsigned short* p = W + (size_t)(n0 + c) * ldw + k0 + hi * 16;
  FragU u;
  u.q[0] = *(const uint4*)p;
  u.q[1] = *(const uint4*)(p + 8);
  return u.v;
}

__device__ __forceinline__ v8f wmma_bf16(v16bf a, v16bf b, v8f c) {
  return __builtin_amdgcn_wmma_f32_16x16x32_bf16(false, a, false, b, (short)0, c,
                                                 false, false);
}

// ---------------------------------------------------------------------------
__global__ void f32_to_bf16_kernel(const float* __restrict__ in,
                                   unsigned short* __restrict__ out, int n) {
  int i = blockIdx.x * blockDim.x + threadIdx.x;
  int stride = gridDim.x * blockDim.x;
  for (; i < n; i += stride) out[i] = f2bf(in[i]);
}

// ---------------------------------------------------------------------------
// QKV GEMM (M=4096 rows of x, N=3072, K=1024) + RoPE epilogue.
// A row-block (16x1024 = 32KB, contiguous) staged once per block via async
// global->LDS; 4 waves x 16x64 output tiles; 32 bf16 WMMA k-steps x4 tiles.
__global__ __launch_bounds__(128) void qkv_rope_kernel(
    const unsigned short* __restrict__ xb,   // [4096,1024] bf16
    const unsigned short* __restrict__ wqb,  // [3072,1024] bf16
    unsigned short* __restrict__ qb,         // [32,2048,64]
    unsigned short* __restrict__ kb,         // [32,2048,64]
    unsigned short* __restrict__ vT)         // [32,64,2048]
{
  __shared__ unsigned short As[16 * 1024];   // 32KB
  int tid = threadIdx.x;
  int wid = tid >> 5;
  int lane = tid & 31;
  int hi = lane >> 4, c = lane & 15;
  int m0 = blockIdx.x * 16;
  int n0 = (blockIdx.y * 4 + wid) * 64;

  {  // async-stage A (contiguous 32KB = 2048 x 16B chunks)
    const unsigned short* gA = xb + (size_t)m0 * 1024;
    for (int i = tid; i < 2048; i += 128)
      async_copy_b128(gA + i * 8, &As[i * 8]);
    wait_async0();
  }
  __syncthreads();

  v8f acc[4] = {};
  for (int k = 0; k < 1024; k += 32) {
    v16bf a = load_A_frag(As, 1024, 0, k);   // from LDS
#pragma unroll
    for (int t = 0; t < 4; ++t)
      acc[t] = wmma_bf16(a, load_B_frag(wqb, 1024, n0 + 16 * t, k), acc[t]);
  }

  int s = n0 >> 10;            // 0=q 1=k 2=v (wave-uniform: n0 is 64-aligned)
  int h = (n0 & 1023) >> 6;
  int bb = m0 >> 11;
  int nseq0 = m0 & 2047;

  if (s == 2) {                // V: store transposed [bh][e][n], packed 16B
#pragma unroll
    for (int t = 0; t < 4; ++t) {
      int e = 16 * t + c;
      union { uint4 q; unsigned short h8[8]; } pk;
#pragma unroll
      for (int r = 0; r < 8; ++r) pk.h8[r] = f2bf(acc[t][r]);
      unsigned short* dst =
          vT + ((size_t)(bb * 16 + h) * 64 + e) * 2048 + nseq0 + hi * 8;
      *(uint4*)dst = pk.q;
    }
  } else {                     // Q/K: RoPE rotate adjacent-column pairs
    unsigned short* dstb =
        ((s == 0) ? qb : kb) + (size_t)(bb * 16 + h) * 2048 * 64;
#pragma unroll
    for (int t = 0; t < 4; ++t) {
      int e = 16 * t + c;
      float inv = __powf(10000.0f, -(float)(e & ~1) * (1.0f / 64.0f));
#pragma unroll
      for (int r = 0; r < 8; ++r) {
        float f = acc[t][r];
        float part = __shfl_xor(f, 1, 32);   // even<->odd column partner
        int n = nseq0 + hi * 8 + r;
        float sv, cv;
        __sincosf((float)n * inv, &sv, &cv);
        float o = (e & 1) ? (part * sv + f * cv)   // xo' = xe*sin + xo*cos
                          : (f * cv - part * sv);  // xe' = xe*cos - xo*sin
        dstb[(size_t)n * 64 + e] = f2bf(o);
      }
    }
  }
}

// ---------------------------------------------------------------------------
// Flash attention: 4 waves per block, all on the SAME (b,h); each wave owns a
// 16-row q tile.  K/V 32-kseq tiles are double-buffered in LDS via async
// global->LDS copies shared by the whole block (4x L2-traffic cut).
// Per step: 4 WMMA for S, online softmax, LDS P-transpose, 4 WMMA for PV.
__global__ __launch_bounds__(128) void attn_kernel(
    const unsigned short* __restrict__ qb,
    const unsigned short* __restrict__ kb,
    const unsigned short* __restrict__ vT,
    unsigned short* __restrict__ ob)  // [4096,1024] bf16 (row b*2048+n, col h*64+e)
{
  __shared__ unsigned short Ks[2][32 * 64];  // 2 x 4KB, [kloc][e] row-major
  __shared__ unsigned short Vs[2][64 * 32];  // 2 x 4KB, [e][kloc] row-major
  __shared__ unsigned short Ps[4][16 * 32];  // per-wave P transpose scratch

  int tid = threadIdx.x;
  int wid = tid >> 5;
  int lane = tid & 31;
  int hi = lane >> 4, c = lane & 15;
  int gw = blockIdx.x * 4 + wid;   // 0..4095
  int qt = gw & 127;
  int bh = gw >> 7;                // 0..31, identical for all 4 waves
  int nq0 = qt * 16;

  const unsigned short* qh = qb + (size_t)bh * 2048 * 64;
  const unsigned short* kh = kb + (size_t)bh * 2048 * 64;
  const unsigned short* vh = vT + (size_t)bh * 64 * 2048;

  v16bf qa0 = load_A_frag(qh, 64, nq0, 0);
  v16bf qa1 = load_A_frag(qh, 64, nq0, 32);

  v8f o0 = {}, o1 = {}, o2 = {}, o3 = {};
  float m[8], l[8];
#pragma unroll
  for (int r = 0; r < 8; ++r) { m[r] = -1e30f; l[r] = 0.0f; }
  const float scale = 0.125f;      // hd^-0.5

  // stage K/V tiles for kseq block kt into LDS buffer `buf`
  auto stage = [&](int buf, int kt) {
    const unsigned short* gk = kh + (size_t)kt * 64;   // contiguous 4KB
    for (int i = tid; i < 256; i += 128)
      async_copy_b128(gk + i * 8, &Ks[buf][i * 8]);
    for (int i = tid; i < 256; i += 128) {             // 64 rows x 64B
      int row = i >> 2, off = (i & 3) * 8;
      async_copy_b128(vh + (size_t)row * 2048 + kt + off,
                      &Vs[buf][row * 32 + off]);
    }
  };

  stage(0, 0);
  for (int it = 0; it < 64; ++it) {
    int kt = it * 32;
    int buf = it & 1;
    wait_async0();        // my wave's copies for `buf` have landed
    __syncthreads();      // all waves' copies done; prior reads of buf done
    if (it + 1 < 64) stage(buf ^ 1, kt + 32);

    // S = Q K^T for 2 x 16 kseq columns (B-frags from LDS K tile)
    v8f s0 = {}, s1 = {};
#pragma unroll
    for (int sub = 0; sub < 2; ++sub) {
#pragma unroll
      for (int es = 0; es < 2; ++es) {
        const unsigned short* p =
            &Ks[buf][(sub * 16 + c) * 64 + es * 32 + hi * 16];
        FragU u;
        u.q[0] = *(const uint4*)p;
        u.q[1] = *(const uint4*)(p + 8);
        if (sub == 0) s0 = wmma_bf16(es == 0 ? qa0 : qa1, u.v, s0);
        else          s1 = wmma_bf16(es == 0 ? qa0 : qa1, u.v, s1);
      }
    }

    float p0[8], p1[8], alpha[8];
#pragma unroll
    for (int r = 0; r < 8; ++r) {
      float a = s0[r] * scale, b = s1[r] * scale;
      float v = fmaxf(a, b);                       // row-max over 32 columns:
      v = fmaxf(v, __shfl_xor(v, 1, 32));          // butterfly over the 16-lane
      v = fmaxf(v, __shfl_xor(v, 2, 32));          // half (rows 0-7 lo / 8-15 hi)
      v = fmaxf(v, __shfl_xor(v, 4, 32));
      v = fmaxf(v, __shfl_xor(v, 8, 32));
      float mn = fmaxf(m[r], v);
      alpha[r] = __expf(m[r] - mn);
      m[r] = mn;
      p0[r] = __expf(a - mn);
      p1[r] = __expf(b - mn);
      float rs = p0[r] + p1[r];
      rs += __shfl_xor(rs, 1, 32);
      rs += __shfl_xor(rs, 2, 32);
      rs += __shfl_xor(rs, 4, 32);
      rs += __shfl_xor(rs, 8, 32);
      l[r] = l[r] * alpha[r] + rs;
    }
#pragma unroll
    for (int r = 0; r < 8; ++r) {
      o0[r] *= alpha[r]; o1[r] *= alpha[r];
      o2[r] *= alpha[r]; o3[r] *= alpha[r];
    }

    // transpose P (C/D layout) -> A layout through wave-local LDS
#pragma unroll
    for (int r = 0; r < 8; ++r) {
      int row = r + hi * 8;
      Ps[wid][row * 32 + c]      = f2bf(p0[r]);
      Ps[wid][row * 32 + c + 16] = f2bf(p1[r]);
    }
    wait_ds0();
    FragU pu;
    const unsigned short* lp = &Ps[wid][c * 32 + hi * 16];
    pu.q[0] = *(const uint4*)lp;
    pu.q[1] = *(const uint4*)(lp + 8);
    v16bf pf = pu.v;

    // O += P V  (B-frags from LDS V tile, [e][kloc] layout)
#pragma unroll
    for (int t = 0; t < 4; ++t) {
      const unsigned short* p = &Vs[buf][(t * 16 + c) * 32 + hi * 16];
      FragU u;
      u.q[0] = *(const uint4*)p;
      u.q[1] = *(const uint4*)(p + 8);
      if (t == 0) o0 = wmma_bf16(pf, u.v, o0);
      if (t == 1) o1 = wmma_bf16(pf, u.v, o1);
      if (t == 2) o2 = wmma_bf16(pf, u.v, o2);
      if (t == 3) o3 = wmma_bf16(pf, u.v, o3);
    }
  }

  int b = bh >> 4, h = bh & 15;
#pragma unroll
  for (int r = 0; r < 8; ++r) {
    int n = nq0 + r + hi * 8;
    size_t rowoff = (size_t)(b * 2048 + n) * 1024 + h * 64;
    float li = 1.0f / l[r];
    ob[rowoff + 0  + c] = f2bf(o0[r] * li);
    ob[rowoff + 16 + c] = f2bf(o1[r] * li);
    ob[rowoff + 32 + c] = f2bf(o2[r] * li);
    ob[rowoff + 48 + c] = f2bf(o3[r] * li);
  }
}

// ---------------------------------------------------------------------------
// Output projection: [4096,1024] @ w_proj^T + b_proj -> f32 d_out.
// A row-block staged via async global->LDS, as in the QKV kernel.
__global__ __launch_bounds__(128) void proj_kernel(
    const unsigned short* __restrict__ ab,   // [4096,1024] bf16
    const unsigned short* __restrict__ wpb,  // [1024,1024] bf16
    const float* __restrict__ bias,
    float* __restrict__ out)                 // [4096,1024] f32
{
  __shared__ unsigned short As[16 * 1024];   // 32KB
  int tid = threadIdx.x;
  int wid = tid >> 5;
  int lane = tid & 31;
  int hi = lane >> 4, c = lane & 15;
  int m0 = blockIdx.x * 16;
  int n0 = (blockIdx.y * 4 + wid) * 64;

  {
    const unsigned short* gA = ab + (size_t)m0 * 1024;
    for (int i = tid; i < 2048; i += 128)
      async_copy_b128(gA + i * 8, &As[i * 8]);
    wait_async0();
  }
  __syncthreads();

  v8f acc[4] = {};
  for (int k = 0; k < 1024; k += 32) {
    v16bf a = load_A_frag(As, 1024, 0, k);
#pragma unroll
    for (int t = 0; t < 4; ++t)
      acc[t] = wmma_bf16(a, load_B_frag(wpb, 1024, n0 + 16 * t, k), acc[t]);
  }
#pragma unroll
  for (int t = 0; t < 4; ++t) {
    float bv = bias[n0 + 16 * t + c];
#pragma unroll
    for (int r = 0; r < 8; ++r) {
      int row = m0 + r + hi * 8;
      out[(size_t)row * 1024 + n0 + 16 * t + c] = acc[t][r] + bv;
    }
  }
}

// ---------------------------------------------------------------------------
extern "C" void kernel_launch(void* const* d_in, const int* in_sizes, int n_in,
                              void* d_out, int out_size, void* d_ws, size_t ws_size,
                              hipStream_t stream) {
  (void)in_sizes; (void)n_in; (void)out_size; (void)ws_size;
  const float* x     = (const float*)d_in[0];   // [2,2048,1024]
  const float* wqkv  = (const float*)d_in[1];   // [3072,1024]
  const float* wproj = (const float*)d_in[2];   // [1024,1024]
  const float* bproj = (const float*)d_in[3];   // [1024]
  float* out = (float*)d_out;

  // Workspace layout (bf16 buffers), total ~48 MB
  char* ws = (char*)d_ws;
  unsigned short* xb  = (unsigned short*)ws;  ws += (size_t)4096 * 1024 * 2;
  unsigned short* wqb = (unsigned short*)ws;  ws += (size_t)3072 * 1024 * 2;
  unsigned short* wpb = (unsigned short*)ws;  ws += (size_t)1024 * 1024 * 2;
  unsigned short* qb  = (unsigned short*)ws;  ws += (size_t)32 * 2048 * 64 * 2;
  unsigned short* kb  = (unsigned short*)ws;  ws += (size_t)32 * 2048 * 64 * 2;
  unsigned short* vT  = (unsigned short*)ws;  ws += (size_t)32 * 64 * 2048 * 2;
  unsigned short* ab  = (unsigned short*)ws;  ws += (size_t)4096 * 1024 * 2;

  f32_to_bf16_kernel<<<1024, 256, 0, stream>>>(x,     xb,  4096 * 1024);
  f32_to_bf16_kernel<<<1024, 256, 0, stream>>>(wqkv,  wqb, 3072 * 1024);
  f32_to_bf16_kernel<<<1024, 256, 0, stream>>>(wproj, wpb, 1024 * 1024);

  qkv_rope_kernel<<<dim3(256, 12), 128, 0, stream>>>(xb, wqb, qb, kb, vT);
  attn_kernel<<<1024, 128, 0, stream>>>(qb, kb, vT, ab);
  proj_kernel<<<dim3(256, 4), 128, 0, stream>>>(ab, wpb, bproj, out);
}
